// Model_46351287058487
// MI455X (gfx1250) — matrix operand order repaired and checked by reading, not compiled, not run on hardware
//
#include <hip/hip_runtime.h>

// ---------------- problem constants (match reference) ----------------
#define SN      3
#define NA      20000
#define FD      1920
#define HD      512
#define NST     16
#define LN_EPS  1e-5f
#define BM      64                     // GEMM M-tile per block
#define NPAD    20224                  // >= NA + 3*63, multiple of 64
#define MT64    (NPAD / BM)            // 316 GEMM blocks

// ---------------- types ----------------
typedef __attribute__((ext_vector_type(16))) __bf16        v16bf;
typedef __attribute__((ext_vector_type(8)))  float         v8f;
typedef __attribute__((ext_vector_type(4)))  unsigned int  v4u;

union Frag { v16bf v; v4u q[2]; };

__device__ __forceinline__ unsigned short f2bf(float f) {
    unsigned int u = __float_as_uint(f);
    u += 0x7fffu + ((u >> 16) & 1u);       // round-to-nearest-even
    return (unsigned short)(u >> 16);
}
__device__ __forceinline__ float bf2f(unsigned short h) {
    return __uint_as_float(((unsigned int)h) << 16);
}

// ---------------- 1) species bucketing (1 wave, deterministic) ----------------
// Buckets are 64-aligned so every BM-row GEMM tile is species-uniform.
__global__ __launch_bounds__(32) void bucket_kernel(
    const int* __restrict__ species,
    int* __restrict__ rowmap,     // [NPAD] slot -> atom (-1 = pad)
    int* __restrict__ tspec)      // [MT64] 64-row tile -> species
{
    const int lane = threadIdx.x;
    for (int i = lane; i < NPAD; i += 32) rowmap[i] = -1;

    int c0 = 0, c1 = 0, c2 = 0;
    for (int i = lane; i < NA; i += 32) {
        int sp = species[i];
        c0 += (sp == 0); c1 += (sp == 1); c2 += (sp == 2);
    }
    for (int off = 16; off; off >>= 1) {
        c0 += __shfl_xor(c0, off, 32);
        c1 += __shfl_xor(c1, off, 32);
        c2 += __shfl_xor(c2, off, 32);
    }
    const int off1 = (c0 + 63) & ~63;
    const int off2 = off1 + ((c1 + 63) & ~63);

    int b0 = 0, b1 = off1, b2 = off2;
    const unsigned int below = (1u << lane) - 1u;
    for (int base = 0; base < NA; base += 32) {
        int i  = base + lane;
        int sp = (i < NA) ? species[i] : -1;
        unsigned int m0 = (unsigned int)__ballot(sp == 0);
        unsigned int m1 = (unsigned int)__ballot(sp == 1);
        unsigned int m2 = (unsigned int)__ballot(sp == 2);
        if (sp == 0) rowmap[b0 + __popc(m0 & below)] = i;
        if (sp == 1) rowmap[b1 + __popc(m1 & below)] = i;
        if (sp == 2) rowmap[b2 + __popc(m2 & below)] = i;
        b0 += __popc(m0); b1 += __popc(m1); b2 += __popc(m2);
    }
    for (int t = lane; t < MT64; t += 32) {
        int slot = t * BM;
        tspec[t] = (slot < off1) ? 0 : ((slot < off2) ? 1 : 2);
    }
}

// ---------------- 2) pack f32 weights -> bf16 B-fragment order ----------------
// Layout: Wp[((s*Kt + kt)*32 + nt)*512 + lane*16 + e]
//   = W[s][kt*32 + (lane>>4)*16 + e][nt*16 + (lane&15)]
__global__ __launch_bounds__(256) void pack_w_kernel(
    const float* __restrict__ W,            // [S, K, 512]
    unsigned short* __restrict__ Wp,
    int K, int ntiles_total)
{
    const int tile = blockIdx.x * 8 + (threadIdx.x >> 5);
    if (tile >= ntiles_total) return;
    const int lane = threadIdx.x & 31;
    const int nt  = tile & 31;
    const int skt = tile >> 5;
    const int Kt  = K >> 5;
    const int s   = skt / Kt;
    const int kt  = skt - s * Kt;
    const float* src = W + (size_t)s * K * HD;
    unsigned short* dst = Wp + (size_t)tile * 512 + lane * 16;
    const int n     = (nt << 4) + (lane & 15);
    const int kbase = (kt << 5) + ((lane >> 4) << 4);
#pragma unroll
    for (int e = 0; e < 16; ++e)
        dst[e] = f2bf(src[(size_t)(kbase + e) * HD + n]);
}

// ---------------- 3) LayerNorm + per-species affine -> compacted bf16 ----------------
__global__ __launch_bounds__(256) void ln_kernel(
    const float* __restrict__ features,
    const float* __restrict__ gamma,
    const float* __restrict__ beta,
    const int* __restrict__ species,
    const int* __restrict__ rowmap,
    unsigned short* __restrict__ Xc)        // [NPAD, FD] bf16
{
    const int slot = blockIdx.x;
    const int tid  = threadIdx.x;
    unsigned short* out = Xc + (size_t)slot * FD;
    const int atom = rowmap[slot];
    if (atom < 0) {
        for (int c = tid; c < FD; c += 256) out[c] = 0;
        return;
    }
    const float* x = features + (size_t)atom * FD;
    float s1 = 0.f, s2 = 0.f;
    for (int c = tid; c < FD; c += 256) { float v = x[c]; s1 += v; s2 += v * v; }
    __shared__ float sh1[256], sh2[256];
    sh1[tid] = s1; sh2[tid] = s2;
    __syncthreads();
    for (int off = 128; off; off >>= 1) {
        if (tid < off) { sh1[tid] += sh1[tid + off]; sh2[tid] += sh2[tid + off]; }
        __syncthreads();
    }
    const float mean = sh1[0] * (1.f / FD);
    const float var  = sh2[0] * (1.f / FD) - mean * mean;
    const float rstd = rsqrtf(var + LN_EPS);
    const int sp = species[atom];
    const float* g  = gamma + (size_t)sp * FD;
    const float* be = beta  + (size_t)sp * FD;
    for (int c = tid; c < FD; c += 256)
        out[c] = f2bf((x[c] - mean) * rstd * g[c] + be[c]);
}

// ---------------- 4) WMMA GEMM + bias + SiLU ----------------
// A: [NPAD, K] bf16 row-major. Block = 64x512 output tile (species-uniform).
// 8 waves; wave owns 4 n-tiles x 4 m-subtiles = 16 WMMA accumulators.
// Per K-step: 8 A-frag b128 + 8 B-frag b128 loads feed 16 WMMAs.
__global__ __launch_bounds__(256) void gemm_silu_kernel(
    const unsigned short* __restrict__ A,
    const unsigned short* __restrict__ Wp,
    const float* __restrict__ bias,         // [S, 512]
    const int* __restrict__ tspec,
    unsigned short* __restrict__ Out,
    int K)
{
    const int m    = blockIdx.x;            // 64-row tile
    const int lane = threadIdx.x & 31;
    const int wave = threadIdx.x >> 5;
    const int s    = tspec[m];
    const int Kt   = K >> 5;
    const int nt0  = wave << 2;

    v8f acc[4][4];                          // [m-subtile][n-tile]
#pragma unroll
    for (int i = 0; i < 4; ++i)
#pragma unroll
        for (int j = 0; j < 4; ++j)
#pragma unroll
            for (int r = 0; r < 8; ++r) acc[i][j][r] = 0.f;

    // A-frag addressing: M = lane&15,
    // elems 0..15 at K = (lane>>4)*16 + e (matches packed B).
    const unsigned short* abase =
        A + (size_t)(m * BM + (lane & 15)) * K + ((lane >> 4) << 4);
    const unsigned short* wbase =
        Wp + (size_t)s * Kt * (32 * 512) + lane * 16;

    for (int kt = 0; kt < Kt; ++kt) {
        Frag fa[4];
#pragma unroll
        for (int i = 0; i < 4; ++i) {
            const unsigned short* ap = abase + (size_t)(i * 16) * K + kt * 32;
            fa[i].q[0] = *(const v4u*)(ap);
            fa[i].q[1] = *(const v4u*)(ap + 8);
        }
        const unsigned short* wk = wbase + (size_t)kt * (32 * 512);
#pragma unroll
        for (int j = 0; j < 4; ++j) {
            Frag fb;
            const unsigned short* wt = wk + (nt0 + j) * 512;
            fb.q[0] = *(const v4u*)(wt);
            fb.q[1] = *(const v4u*)(wt + 8);
#pragma unroll
            for (int i = 0; i < 4; ++i)
                acc[i][j] = __builtin_amdgcn_wmma_f32_16x16x32_bf16(
                    false, fa[i].v, false, fb.v, (short)0, acc[i][j],
                    false, false);
        }
    }

    // C/D layout: VGPR r -> M = r + (lane>>4)*8, N = lane&15.
    const int Mhi = (lane >> 4) << 3;
#pragma unroll
    for (int j = 0; j < 4; ++j) {
        const int col = ((nt0 + j) << 4) + (lane & 15);
        const float b = bias[s * HD + col];
#pragma unroll
        for (int i = 0; i < 4; ++i) {
            unsigned short* orow = Out + (size_t)(m * BM + i * 16) * HD + col;
#pragma unroll
            for (int r = 0; r < 8; ++r) {
                float x = acc[i][j][r] + b;
                float y = x / (1.f + __expf(-x));      // SiLU
                orow[(size_t)(r + Mhi) * HD] = f2bf(y);
            }
        }
    }
}

// ---------------- 5) per-row dot with w3 + b3 + comp_w -> e[atom] ----------------
__global__ __launch_bounds__(256) void dot_kernel(
    const unsigned short* __restrict__ H2,  // [NPAD, HD] bf16
    const float* __restrict__ w3,           // [S, HD]
    const float* __restrict__ b3,           // [S]
    const float* __restrict__ comp_w,       // [S]
    const int* __restrict__ rowmap,
    const int* __restrict__ species,
    float* __restrict__ e_arr)              // [NA]
{
    const int r = blockIdx.x * 8 + (threadIdx.x >> 5);
    const int lane = threadIdx.x & 31;
    if (r >= NPAD) return;
    const int atom = rowmap[r];
    if (atom < 0) return;
    const int sp = species[atom];
    const unsigned short* h = H2 + (size_t)r * HD;
    const float* w = w3 + (size_t)sp * HD;
    float acc = 0.f;
#pragma unroll
    for (int e = 0; e < 16; ++e) {
        int col = lane + (e << 5);
        acc += bf2f(h[col]) * w[col];
    }
    for (int off = 16; off; off >>= 1) acc += __shfl_xor(acc, off, 32);
    if (lane == 0) e_arr[atom] = acc + b3[sp] + comp_w[sp];
}

// ---------------- 6) deterministic segment sum ----------------
__global__ __launch_bounds__(256) void reduce_kernel(
    const float* __restrict__ e_arr,
    const int* __restrict__ struct_id,
    float* __restrict__ out)
{
    const int b = blockIdx.x, tid = threadIdx.x;
    float acc = 0.f;
    for (int i = tid; i < NA; i += 256)
        if (struct_id[i] == b) acc += e_arr[i];
    __shared__ float sh[256];
    sh[tid] = acc; __syncthreads();
    for (int off = 128; off; off >>= 1) {
        if (tid < off) sh[tid] += sh[tid + off];
        __syncthreads();
    }
    if (tid == 0) out[b] = sh[0];
}

// ---------------- host launcher ----------------
extern "C" void kernel_launch(void* const* d_in, const int* in_sizes, int n_in,
                              void* d_out, int out_size, void* d_ws, size_t ws_size,
                              hipStream_t stream)
{
    const float* features = (const float*)d_in[0];
    const float* ln_gamma = (const float*)d_in[1];
    const float* ln_beta  = (const float*)d_in[2];
    const float* W1       = (const float*)d_in[3];
    const float* b1       = (const float*)d_in[4];
    const float* W2       = (const float*)d_in[5];
    const float* b2       = (const float*)d_in[6];
    const float* w3       = (const float*)d_in[7];
    const float* b3       = (const float*)d_in[8];
    const float* comp_w   = (const float*)d_in[9];
    const int*   species  = (const int*)d_in[10];
    const int*   struct_id= (const int*)d_in[11];
    float* out = (float*)d_out;

    char* p = (char*)d_ws;
    auto alloc = [&](size_t bytes) -> char* {
        char* r = p; p += (bytes + 255) & ~(size_t)255; return r;
    };
    int*            rowmap = (int*)           alloc((size_t)NPAD * 4);
    int*            tspec  = (int*)           alloc((size_t)MT64 * 4);
    float*          e_arr  = (float*)         alloc((size_t)NA * 4);
    unsigned short* W1p    = (unsigned short*)alloc((size_t)SN * FD * HD * 2);
    unsigned short* W2p    = (unsigned short*)alloc((size_t)SN * HD * HD * 2);
    unsigned short* H1     = (unsigned short*)alloc((size_t)NPAD * HD * 2);
    unsigned short* Xc     = (unsigned short*)alloc((size_t)NPAD * FD * 2);
    unsigned short* H2b    = Xc;   // alias: Xc dead after GEMM1, reuse for H2
    if ((size_t)(p - (char*)d_ws) > ws_size) return;   // workspace too small

    bucket_kernel<<<1, 32, 0, stream>>>(species, rowmap, tspec);

    const int nt1 = SN * (FD / 32) * 32;   // 5760 W1 tiles
    const int nt2 = SN * (HD / 32) * 32;   // 1536 W2 tiles
    pack_w_kernel<<<(nt1 + 7) / 8, 256, 0, stream>>>(W1, W1p, FD, nt1);
    pack_w_kernel<<<(nt2 + 7) / 8, 256, 0, stream>>>(W2, W2p, HD, nt2);

    ln_kernel<<<NPAD, 256, 0, stream>>>(features, ln_gamma, ln_beta,
                                        species, rowmap, Xc);

    gemm_silu_kernel<<<MT64, 256, 0, stream>>>(Xc, W1p, b1, tspec, H1, FD);
    gemm_silu_kernel<<<MT64, 256, 0, stream>>>(H1, W2p, b2, tspec, H2b, HD);

    dot_kernel<<<(NPAD + 7) / 8, 256, 0, stream>>>(H2b, w3, b3, comp_w,
                                                   rowmap, species, e_arr);
    reduce_kernel<<<NST, 256, 0, stream>>>(e_arr, struct_id, out);
}
